// CryptoRWKV_TimeMix_13039520710708
// MI455X (gfx1250) — matrix operand order repaired and checked by reading, not compiled, not run on hardware
//
#include <hip/hip_runtime.h>
#include <hip/hip_bf16.h>
#include <cstdint>

// ---------------------------------------------------------------------------
// RWKV time-mix for MI455X (gfx1250, wave32, WMMA bf16, async global->LDS,
// double-buffered LDS pipeline in the GEMMs)
// ---------------------------------------------------------------------------

typedef __attribute__((ext_vector_type(16))) __bf16 v16bf;
typedef __attribute__((ext_vector_type(8)))  float  v8f;
typedef __attribute__((__vector_size__(16))) int    v4i;

#define DEV __device__ __forceinline__
#define AS1 __attribute__((address_space(1)))
#define AS3 __attribute__((address_space(3)))

constexpr int Bc = 8, Tc = 2048, Cc = 1024, Hc = 16, Nh = 64, CK = 256, NC = 8;
constexpr int Mrows = Bc * Tc; // 16384

#if __has_builtin(__builtin_amdgcn_global_load_async_to_lds_b128)
#define HAVE_ASYNC_LDS 1
#endif

DEV AS1 v4i* gcast(const void* p) {
  return (AS1 v4i*)(unsigned long long)(size_t)p;
}
DEV AS3 v4i* lcast(const void* p) {
  return (AS3 v4i*)(unsigned)(size_t)p;
}
DEV void wait_async0() {
#if __has_builtin(__builtin_amdgcn_s_wait_asynccnt)
  __builtin_amdgcn_s_wait_asynccnt(0);
#else
  asm volatile("s_wait_asynccnt 0" ::: "memory");
#endif
}

DEV __bf16 f2bf(float f) {
  unsigned u = __builtin_bit_cast(unsigned, f);
  u += 0x7FFFu + ((u >> 16) & 1u);            // round-to-nearest-even
  unsigned short s = (unsigned short)(u >> 16);
  return __builtin_bit_cast(__bf16, s);
}

// A-matrix fragment, 16x32 bf16, row-major source with leading dim ld.
// Layout: lane L<16 -> row L, K {0..15}; lane L>=16 -> row L-16, K {16..31}
DEV v16bf load_frag_a(const __bf16* __restrict__ src, int ld) {
  const int lane = threadIdx.x & 31;
  const int row  = lane & 15;
  const int kh   = lane >> 4;
  const __bf16* p = src + row * ld;
  v16bf f;
#pragma unroll
  for (int h = 0; h < 16; ++h) {
    int k = (kh << 4) + h;
    f[h] = p[k];
  }
  return f;
}

// B-matrix fragment, 32x16 bf16, where B[k][n] = src[n*ld + k] (column-contiguous).
DEV v16bf load_frag_b(const __bf16* __restrict__ src, int ld) {
  const int lane = threadIdx.x & 31;
  const int col  = lane & 15;
  const int kh   = lane >> 4;
  const __bf16* p = src + col * ld + kh * 16;
  v16bf f;
#pragma unroll
  for (int h = 0; h < 16; ++h) f[h] = p[h];
  return f;
}

// C/D 16x16 f32: lane 0-15 -> N=lane, M=vgpr; lane 16-31 -> N=lane-16, M=vgpr+8.
DEV v8f load_frag_c(const float* __restrict__ src, int ld) {
  const int lane = threadIdx.x & 31;
  const int col  = lane & 15;
  const int rb   = (lane >> 4) * 8;
  v8f c;
#pragma unroll
  for (int i = 0; i < 8; ++i) c[i] = src[(rb + i) * ld + col];
  return c;
}

DEV void store_frag_c(float* __restrict__ dst, int ld, v8f c) {
  const int lane = threadIdx.x & 31;
  const int col  = lane & 15;
  const int rb   = (lane >> 4) * 8;
#pragma unroll
  for (int i = 0; i < 8; ++i) dst[(rb + i) * ld + col] = c[i];
}

DEV v8f wmma_bf(v16bf a, v16bf b, v8f c) {
  return __builtin_amdgcn_wmma_f32_16x16x32_bf16(false, a, false, b, (short)0, c,
                                                 false, false);
}

// ---------------------------------------------------------------------------
// Kernel: f32 -> bf16 weight conversion
// ---------------------------------------------------------------------------
__global__ __launch_bounds__(256) void cvt_bf16_kernel(const float* __restrict__ s,
                                                       __bf16* __restrict__ d, int n) {
  for (int i = blockIdx.x * blockDim.x + threadIdx.x; i < n;
       i += gridDim.x * blockDim.x)
    d[i] = f2bf(s[i]);
}

// ---------------------------------------------------------------------------
// Kernel: volatility gate + time-shift mix -> xk/xv/xr (bf16)
// ---------------------------------------------------------------------------
__global__ __launch_bounds__(256) void prep_kernel(
    const float* __restrict__ x, const float* __restrict__ vol,
    const float* __restrict__ Wvol, const float* __restrict__ bvol,
    const float* __restrict__ tmk, const float* __restrict__ tmv,
    const float* __restrict__ tmr, __bf16* __restrict__ xk,
    __bf16* __restrict__ xv, __bf16* __restrict__ xr) {
  int idx = blockIdx.x * blockDim.x + threadIdx.x;   // < B*T*C = 16M
  int c  = idx & (Cc - 1);
  int bt = idx >> 10;                                 // b*T + t
  int t  = bt & (Tc - 1);
  float wv = Wvol[c], bv = bvol[c];
  float ve = 1.0f / (1.0f + __expf(-(vol[bt] * wv + bv)));
  float xg = x[idx] * ve;
  float xp = 0.0f;
  if (t > 0) {
    float vep = 1.0f / (1.0f + __expf(-(vol[bt - 1] * wv + bv)));
    xp = x[idx - Cc] * vep;
  }
  float xx = xp - xg;
  xk[idx] = f2bf(xg + xx * tmk[c]);
  xv[idx] = f2bf(xg + xx * tmv[c]);
  xr[idx] = f2bf(xg + xx * tmr[c]);
}

// ---------------------------------------------------------------------------
// Kernel: Y = X @ W^T   (X: MxK bf16 row-major, W: NxK bf16 row-major)
// Block tile 128x128, K-tile 64, DOUBLE-BUFFERED async global->LDS pipeline:
//   iter i: issue asyncs for tile i+1 into nxt, 16 WMMAs on cur,
//           s_wait_asynccnt 0, one barrier.
// ---------------------------------------------------------------------------
__global__ __launch_bounds__(256) void gemm_bf16_kernel(
    const __bf16* __restrict__ X, const __bf16* __restrict__ W,
    float* __restrict__ Yf, __bf16* __restrict__ Yb, int M, int N, int K) {
  __shared__ __bf16 As[2][128 * 64];   // 2 x 16 KB
  __shared__ __bf16 Bs[2][128 * 64];   // 2 x 16 KB

  const int tid  = threadIdx.x;
  const int bm   = blockIdx.y * 128;
  const int bn   = blockIdx.x * 128;
  const int lrow = tid >> 1;           // 0..127
  const int lcol = (tid & 1) * 32;     // 0 / 32 (64B segment)
  const int wave = tid >> 5;
  const int wm   = wave >> 2;          // 0..1  -> 64-row half
  const int wn   = wave & 3;           // 0..3  -> 32-col quarter

  v8f acc[4][2] = {};

  const __bf16* xrow = X + (size_t)(bm + lrow) * K + lcol;
  const __bf16* wrow = W + (size_t)(bn + lrow) * K + lcol;

#ifdef HAVE_ASYNC_LDS
  auto stage = [&](int k0, int buf) {
    const __bf16* gx = xrow + k0;
    const __bf16* gw = wrow + k0;
    __bf16* lx = As[buf] + lrow * 64 + lcol;
    __bf16* lw = Bs[buf] + lrow * 64 + lcol;
#pragma unroll
    for (int j = 0; j < 4; ++j) {
      __builtin_amdgcn_global_load_async_to_lds_b128(gcast(gx + j * 8),
                                                     lcast(lx + j * 8), 0, 0);
      __builtin_amdgcn_global_load_async_to_lds_b128(gcast(gw + j * 8),
                                                     lcast(lw + j * 8), 0, 0);
    }
  };
#else
  auto stage = [&](int k0, int buf) {
    const uint4* sa = (const uint4*)(xrow + k0);
    const uint4* sb = (const uint4*)(wrow + k0);
    uint4* da = (uint4*)(As[buf] + lrow * 64 + lcol);
    uint4* db = (uint4*)(Bs[buf] + lrow * 64 + lcol);
#pragma unroll
    for (int j = 0; j < 4; ++j) { da[j] = sa[j]; db[j] = sb[j]; }
  };
#endif

  const int nb = K / 64;
  stage(0, 0);
  wait_async0();
  __syncthreads();

  for (int it = 0; it < nb; ++it) {
    const int cur = it & 1;
    const int nxt = cur ^ 1;
    if (it + 1 < nb) stage((it + 1) * 64, nxt);   // overlap DMA with WMMA

#pragma unroll
    for (int ks2 = 0; ks2 < 64; ks2 += 32) {
      v16bf a[4], b[2];
#pragma unroll
      for (int i = 0; i < 4; ++i)
        a[i] = load_frag_a(As[cur] + (wm * 64 + i * 16) * 64 + ks2, 64);
#pragma unroll
      for (int j = 0; j < 2; ++j)
        b[j] = load_frag_b(Bs[cur] + (wn * 32 + j * 16) * 64 + ks2, 64);
#pragma unroll
      for (int i = 0; i < 4; ++i)
#pragma unroll
        for (int j = 0; j < 2; ++j) acc[i][j] = wmma_bf(a[i], b[j], acc[i][j]);
    }

    wait_async0();      // next tile landed
    __syncthreads();    // publish nxt; all waves done reading cur
  }

  const int lane = tid & 31;
  const int col  = lane & 15;
  const int rb   = (lane >> 4) * 8;
#pragma unroll
  for (int i = 0; i < 4; ++i)
#pragma unroll
    for (int j = 0; j < 2; ++j) {
      int gm = bm + wm * 64 + i * 16 + rb;
      int gn = bn + wn * 32 + j * 16 + col;
      if (Yf) {
#pragma unroll
        for (int e = 0; e < 8; ++e)
          Yf[(size_t)(gm + e) * N + gn] = acc[i][j][e];
      } else {
#pragma unroll
        for (int e = 0; e < 8; ++e)
          Yb[(size_t)(gm + e) * N + gn] = f2bf(acc[i][j][e]);
      }
    }
}

// ---------------------------------------------------------------------------
// Kernel: chunked linear attention + fused GroupNorm.
// One block per (b,h). 8 waves; wave handles 32 t-rows of each 256-row chunk.
// Dynamic LDS carve (136 KB):
//   [0,32K)    ks  : k chunk row-major 256x64            (overlaid by ysc later)
//   [32K,64K)  kts : k^T 64x256                          (overlaid by ysc later)
//   [64K,96K)  vts : v^T 64x256
//   [96K,112K) stf : state f32 64x64 (row n major)
//   [112K,120K)stb : state bf16 TRANSPOSED 64x64 (row m major)
//   [120K,136K)atts: per-(wave,rt) 16x32 bf16 att scratch
// ---------------------------------------------------------------------------
__global__ __launch_bounds__(256) void attn_kernel(
    const __bf16* __restrict__ rP, const __bf16* __restrict__ kP,
    const __bf16* __restrict__ vP, const float* __restrict__ td,
    const float* __restrict__ gnw, const float* __restrict__ gnb,
    __bf16* __restrict__ yout) {
  extern __shared__ char smem[];
  __bf16* ks   = (__bf16*)(smem);
  __bf16* kts  = (__bf16*)(smem + 32768);
  __bf16* vts  = (__bf16*)(smem + 65536);
  float*  stf  = (float*)(smem + 98304);
  __bf16* stb  = (__bf16*)(smem + 114688);
  __bf16* atts = (__bf16*)(smem + 122880);
  float*  ysc  = (float*)(smem);   // overlay of ks+kts: 8 waves * 2048 f32

  const int bh = blockIdx.x;
  const int b  = bh / Hc;
  const int h  = bh % Hc;
  const int tid  = threadIdx.x;
  const int wave = tid >> 5;
  const int lane = tid & 31;
  const int col  = lane & 15;
  const int rb8  = (lane >> 4) * 8;
  const float w  = __expf(-__expf(td[h]));

  // zero state
  for (int i = tid; i < Nh * Nh; i += 256) { stf[i] = 0.0f; stb[i] = f2bf(0.0f); }
  __syncthreads();

  for (int ci = 0; ci < NC; ++ci) {
    const size_t chbase = (size_t)(b * Tc + ci * CK) * Cc + h * Nh;

    // ---- stage k (row + transposed) and v (transposed) : thread = row ----
    {
      const uint4* s4 = (const uint4*)(kP + chbase + (size_t)tid * Cc);
      uint4* d4 = (uint4*)(ks + tid * 64);
#pragma unroll
      for (int j = 0; j < 8; ++j) {
        uint4 q = s4[j];
        d4[j] = q;
        unsigned vv[4] = {q.x, q.y, q.z, q.w};
#pragma unroll
        for (int e = 0; e < 4; ++e) {
          kts[(j * 8 + e * 2 + 0) * CK + tid] =
              __builtin_bit_cast(__bf16, (unsigned short)(vv[e] & 0xFFFFu));
          kts[(j * 8 + e * 2 + 1) * CK + tid] =
              __builtin_bit_cast(__bf16, (unsigned short)(vv[e] >> 16));
        }
      }
      const uint4* sv = (const uint4*)(vP + chbase + (size_t)tid * Cc);
#pragma unroll
      for (int j = 0; j < 8; ++j) {
        uint4 q = sv[j];
        unsigned vv[4] = {q.x, q.y, q.z, q.w};
#pragma unroll
        for (int e = 0; e < 4; ++e) {
          vts[(j * 8 + e * 2 + 0) * CK + tid] =
              __builtin_bit_cast(__bf16, (unsigned short)(vv[e] & 0xFFFFu));
          vts[(j * 8 + e * 2 + 1) * CK + tid] =
              __builtin_bit_cast(__bf16, (unsigned short)(vv[e] >> 16));
        }
      }
    }
    __syncthreads();

    // ---- y = w * (att @ v + r @ state_old) for rows t0..t0+31 ----
    const int t0 = wave * 32;
    v16bf ra[2][2];
#pragma unroll
    for (int rt = 0; rt < 2; ++rt)
#pragma unroll
      for (int g = 0; g < 2; ++g)
        ra[rt][g] = load_frag_a(rP + chbase + (size_t)(t0 + rt * 16) * Cc + g * 32, Cc);

    v8f yacc[2][4] = {};
    // r @ state_old  (state^T bf16, column-contiguous for B fragments)
#pragma unroll
    for (int rt = 0; rt < 2; ++rt)
#pragma unroll
      for (int ct = 0; ct < 4; ++ct)
#pragma unroll
        for (int g = 0; g < 2; ++g)
          yacc[rt][ct] = wmma_bf(ra[rt][g],
                                 load_frag_b(stb + ct * 16 * 64 + g * 32, 64),
                                 yacc[rt][ct]);
    // att @ v with per-wave LDS bounce of att fragments
    for (int s0 = 0; s0 < CK; s0 += 32) {
#pragma unroll
      for (int rt = 0; rt < 2; ++rt) {
        v8f at[2] = {};
#pragma unroll
        for (int g = 0; g < 2; ++g)
#pragma unroll
          for (int st = 0; st < 2; ++st)
            at[st] = wmma_bf(ra[rt][g],
                             load_frag_b(ks + (s0 + st * 16) * 64 + g * 32, 64),
                             at[st]);
        __bf16* aw = atts + (wave * 2 + rt) * 512;
#pragma unroll
        for (int st = 0; st < 2; ++st)
#pragma unroll
          for (int e = 0; e < 8; ++e)
            aw[(rb8 + e) * 32 + st * 16 + col] = f2bf(at[st][e]);
      }
      asm volatile("s_wait_dscnt 0" ::: "memory");   // wave-local LDS RAW
#pragma unroll
      for (int rt = 0; rt < 2; ++rt) {
        v16bf af = load_frag_a(atts + (wave * 2 + rt) * 512, 32);
#pragma unroll
        for (int ct = 0; ct < 4; ++ct)
          yacc[rt][ct] = wmma_bf(af, load_frag_b(vts + ct * 16 * CK + s0, CK),
                                 yacc[rt][ct]);
      }
      asm volatile("" ::: "memory");
    }
#pragma unroll
    for (int rt = 0; rt < 2; ++rt)
#pragma unroll
      for (int ct = 0; ct < 4; ++ct)
#pragma unroll
        for (int e = 0; e < 8; ++e) yacc[rt][ct][e] *= w;

    // ---- state_new = w * (state_old + k^T @ v): wave owns 2 of 16 tiles ----
    v8f sacc[2];
#pragma unroll
    for (int u = 0; u < 2; ++u) {
      int tt = wave * 2 + u, nt = tt >> 2, mt = tt & 3;
      sacc[u] = load_frag_c(stf + nt * 16 * 64 + mt * 16, 64);
      for (int s0 = 0; s0 < CK; s0 += 32)
        sacc[u] = wmma_bf(load_frag_a(kts + nt * 16 * CK + s0, CK),
                          load_frag_b(vts + mt * 16 * CK + s0, CK), sacc[u]);
#pragma unroll
      for (int e = 0; e < 8; ++e) sacc[u][e] *= w;
    }
    __syncthreads();   // everyone done reading state_old / ks / kts

    // write back state (f32 row-n-major + bf16 transposed)
#pragma unroll
    for (int u = 0; u < 2; ++u) {
      int tt = wave * 2 + u, nt = tt >> 2, mt = tt & 3;
      store_frag_c(stf + nt * 16 * 64 + mt * 16, 64, sacc[u]);
#pragma unroll
      for (int e = 0; e < 8; ++e)
        stb[(mt * 16 + col) * 64 + nt * 16 + rb8 + e] = f2bf(sacc[u][e]);
    }

    // ---- fused GroupNorm over Nh per row; write y (bf16, (B*T,C)) ----
    float* yw = ysc + wave * 2048;   // 32 rows x 64 (overlays ks/kts)
#pragma unroll
    for (int rt = 0; rt < 2; ++rt)
#pragma unroll
      for (int ct = 0; ct < 4; ++ct)
        store_frag_c(yw + rt * 16 * 64 + ct * 16, 64, yacc[rt][ct]);
    asm volatile("s_wait_dscnt 0" ::: "memory");

    {
      const float* myrow = yw + lane * 64;
      float mu = 0.0f;
#pragma unroll
      for (int n = 0; n < Nh; ++n) mu += myrow[n];
      mu *= (1.0f / Nh);
      float var = 0.0f;
#pragma unroll
      for (int n = 0; n < Nh; ++n) { float d = myrow[n] - mu; var += d * d; }
      var *= (1.0f / Nh);
      float inv = rsqrtf(var + 1e-5f);
      int trow = ci * CK + wave * 32 + lane;
      __bf16* orow = yout + (size_t)(b * Tc + trow) * Cc + h * Nh;
#pragma unroll
      for (int n = 0; n < Nh; ++n)
        orow[n] = f2bf((myrow[n] - mu) * inv * gnw[h * Nh + n] + gnb[h * Nh + n]);
    }
    __syncthreads();   // protect ysc/ks/kts before next chunk staging
  }
}

// ---------------------------------------------------------------------------
// Host-side launcher
// ---------------------------------------------------------------------------
extern "C" void kernel_launch(void* const* d_in, const int* in_sizes, int n_in,
                              void* d_out, int out_size, void* d_ws, size_t ws_size,
                              hipStream_t stream) {
  const float* x    = (const float*)d_in[0];
  const float* vol  = (const float*)d_in[1];
  const float* Wvol = (const float*)d_in[2];
  const float* bvol = (const float*)d_in[3];
  const float* tmk  = (const float*)d_in[4];
  const float* tmv  = (const float*)d_in[5];
  const float* tmr  = (const float*)d_in[6];
  const float* td   = (const float*)d_in[7];
  const float* Wk   = (const float*)d_in[8];
  const float* Wv   = (const float*)d_in[9];
  const float* Wr   = (const float*)d_in[10];
  const float* Wo   = (const float*)d_in[11];
  const float* gnw  = (const float*)d_in[12];
  const float* gnb  = (const float*)d_in[13];
  float* out = (float*)d_out;

  char* ws = (char*)d_ws;
  const size_t MB = 1024 * 1024;
  __bf16* xr  = (__bf16*)(ws + 0 * MB);
  __bf16* xk  = (__bf16*)(ws + 32 * MB);
  __bf16* xv  = (__bf16*)(ws + 64 * MB);
  __bf16* rbf = (__bf16*)(ws + 96 * MB);
  __bf16* kbf = (__bf16*)(ws + 128 * MB);
  __bf16* vbf = (__bf16*)(ws + 160 * MB);
  __bf16* wkb = (__bf16*)(ws + 192 * MB);
  __bf16* wvb = (__bf16*)(ws + 194 * MB);
  __bf16* wrb = (__bf16*)(ws + 196 * MB);
  __bf16* wob = (__bf16*)(ws + 198 * MB);
  __bf16* ybf = xr;   // xr dead after r-GEMM -> reuse for y

  // 1) weights -> bf16
  cvt_bf16_kernel<<<1024, 256, 0, stream>>>(Wk, wkb, Cc * Cc);
  cvt_bf16_kernel<<<1024, 256, 0, stream>>>(Wv, wvb, Cc * Cc);
  cvt_bf16_kernel<<<1024, 256, 0, stream>>>(Wr, wrb, Cc * Cc);
  cvt_bf16_kernel<<<1024, 256, 0, stream>>>(Wo, wob, Cc * Cc);

  // 2) gate + time-shift mix
  prep_kernel<<<(Bc * Tc * Cc) / 256, 256, 0, stream>>>(x, vol, Wvol, bvol, tmk,
                                                        tmv, tmr, xk, xv, xr);

  // 3) projections r/k/v (bf16 outputs)
  dim3 ggrid(Cc / 128, Mrows / 128);
  gemm_bf16_kernel<<<ggrid, 256, 0, stream>>>(xr, wrb, nullptr, rbf, Mrows, Cc, Cc);
  gemm_bf16_kernel<<<ggrid, 256, 0, stream>>>(xk, wkb, nullptr, kbf, Mrows, Cc, Cc);
  gemm_bf16_kernel<<<ggrid, 256, 0, stream>>>(xv, wvb, nullptr, vbf, Mrows, Cc, Cc);

  // 4) chunked linear attention + fused GroupNorm
  attn_kernel<<<Bc * Hc, 256, 139264, stream>>>(rbf, kbf, vbf, td, gnw, gnb, ybf);

  // 5) output projection (f32 out)
  gemm_bf16_kernel<<<ggrid, 256, 0, stream>>>(ybf, wob, out, nullptr, Mrows, Cc, Cc);

  (void)in_sizes; (void)n_in; (void)out_size; (void)ws_size;
}
